// SentenceSwitchMoeBlock_44667659878788
// MI455X (gfx1250) — matrix operand order repaired and checked by reading, not compiled, not run on hardware
//
#include <hip/hip_runtime.h>
#include <hip/hip_bf16.h>
#include <stdint.h>

// Problem sizes (fixed by reference)
#define B_ 32
#define S_ 512
#define H_ 768
#define E_ 8
#define F_ 3072

typedef __attribute__((ext_vector_type(16))) __bf16        v16bf;
typedef __attribute__((ext_vector_type(8)))  float         v8f;
typedef __attribute__((ext_vector_type(8)))  unsigned int  v8u;

union FragCast { v8u u; v16bf b; };

// f32x2 -> packed bf16x2 (RNE). Prefer the hardware pack instruction.
__device__ __forceinline__ unsigned cvt_pk_bf16(float lo, float hi) {
#if __has_builtin(__builtin_amdgcn_cvt_pk_bf16_f32)
  auto r = __builtin_amdgcn_cvt_pk_bf16_f32(lo, hi);
  unsigned u;
  __builtin_memcpy(&u, &r, sizeof(u));
  return u;
#else
  union { __bf16 h[2]; unsigned u; } r;
  r.h[0] = (__bf16)lo;
  r.h[1] = (__bf16)hi;
  return r.u;
#endif
}

__device__ __forceinline__ unsigned short f2bf(float f) {
#if __has_builtin(__builtin_amdgcn_cvt_pk_bf16_f32)
  return (unsigned short)(cvt_pk_bf16(f, 0.0f) & 0xFFFFu);
#else
  union { __bf16 h; unsigned short u; } r;
  r.h = (__bf16)f;
  return r.u;
#endif
}

__device__ __forceinline__ float fast_tanh(float x) {
#if __has_builtin(__builtin_amdgcn_tanhf)
  return __builtin_amdgcn_tanhf(x);
#else
  return tanhf(x);
#endif
}

__device__ __forceinline__ float gelu_tanh(float x) {
  float u = 0.7978845608028654f * (x + 0.044715f * x * x * x);
  return 0.5f * x * (1.0f + fast_tanh(u));
}

// Async global->LDS copy (CDNA5): 16B per lane, tracked on ASYNCcnt.
__device__ __forceinline__ void async_copy_b128(unsigned lds_off, const void* gptr) {
  asm volatile("global_load_async_to_lds_b128 %0, %1, off"
               :: "v"(lds_off), "v"(gptr)
               : "memory");
}
__device__ __forceinline__ void wait_asynccnt0() {
  asm volatile("s_wait_asynccnt 0x0" ::: "memory");
}

// ---------------------------------------------------------------------------
// Router: pooled logits + argmax expert choice per batch.
// grid = B_, block = 256
// ---------------------------------------------------------------------------
__global__ void router_kernel(const float* __restrict__ x,
                              const float* __restrict__ gw,
                              float* __restrict__ logits_out,   // [B_,E_]
                              int* __restrict__ choice) {       // [B_]
  const int b = blockIdx.x;
  const int t = threadIdx.x;            // 0..255, 3 H-columns per thread
  __shared__ float pooled[H_];
  __shared__ float lg[E_];

  const float* xb = x + (size_t)b * S_ * H_;
  float a0 = 0.f, a1 = 0.f, a2 = 0.f;
  for (int s = 0; s < S_; ++s) {
    const float* row = xb + (size_t)s * H_;
    a0 += row[t];
    a1 += row[t + 256];
    a2 += row[t + 512];
  }
  const float inv = 1.0f / (float)S_;
  pooled[t]       = a0 * inv;
  pooled[t + 256] = a1 * inv;
  pooled[t + 512] = a2 * inv;
  if (t < E_) lg[t] = 0.f;
  __syncthreads();

  #pragma unroll
  for (int e = 0; e < E_; ++e) {
    const float* g = gw + (size_t)e * H_;
    float p = pooled[t] * g[t] + pooled[t + 256] * g[t + 256] +
              pooled[t + 512] * g[t + 512];
    atomicAdd(&lg[e], p);
  }
  __syncthreads();

  if (t == 0) {
    int best = 0;
    float bv = lg[0];
    #pragma unroll
    for (int e = 1; e < E_; ++e) {
      if (lg[e] > bv) { bv = lg[e]; best = e; }   // first-max tie-break
    }
    choice[b] = best;
    #pragma unroll
    for (int e = 0; e < E_; ++e) logits_out[(size_t)b * E_ + e] = lg[e];
  }
}

// ---------------------------------------------------------------------------
// GEMM tiling: 128x128 block tile, 8 waves (2Mx4N), wave tile 64x32 ->
// 4x2 fragments of 16x16, K-step 32 (v_wmma_f32_16x16x32_bf16).
// LDS: A row-major bf16 [128][32] padded to stride 40 halfwords,
//      B pair-packed u32 [16][128] padded to stride 132 dwords
//      (dword at [k/2][n] = {bf16 B[2k,n] lo, B[2k+1,n] hi} matching the
//       ISA B-fragment layout: VGPR v, half h <-> K=8h+2v,+1 (v<4) / 8+8h+2v,+1 (v>=4); lane<->N).
// ---------------------------------------------------------------------------
#define A_STRIDE 40
#define B_STRIDE 132

// GEMM1: hidden[b] (f32 [S_,H_]) x w1[e] (f32 [H_,F_]) -> gelu(+b1) bf16 -> hbuf
__global__ void gemm1_kernel(const float* __restrict__ x,
                             const float* __restrict__ w1,
                             const float* __restrict__ b1,
                             const int* __restrict__ choice,
                             unsigned short* __restrict__ hbuf) {
  __shared__ unsigned short As[128 * A_STRIDE];
  __shared__ unsigned       Bs[16 * B_STRIDE];

  const int b = blockIdx.z;
  const int e = choice[b];
  const int mBlock = blockIdx.x * 128;   // over S_
  const int nBlock = blockIdx.y * 128;   // over F_
  const int tid  = threadIdx.x;
  const int lane = tid & 31;
  const int l15  = lane & 15;
  const int half16 = (lane < 16) ? 0 : 1;
  const int wid  = tid >> 5;
  const int waveM = (wid & 1) * 64;
  const int waveN = (wid >> 1) * 32;

  const float* Ab  = x  + (size_t)b * S_ * H_;
  const float* Wb  = w1 + (size_t)e * H_ * F_;
  const float* b1e = b1 + (size_t)e * F_;

  v8f acc[4][2];
  #pragma unroll
  for (int mf = 0; mf < 4; ++mf)
    #pragma unroll
    for (int nf = 0; nf < 2; ++nf)
      acc[mf][nf] = (v8f)0.0f;

  for (int k0 = 0; k0 < H_; k0 += 32) {
    __syncthreads();
    // ---- stage A: 128x32 f32 -> bf16 LDS (hardware pack converts)
    #pragma unroll
    for (int i = 0; i < 4; ++i) {
      int idx = tid + i * 256;           // 1024 float4 chunks
      int row = idx >> 3, c4 = idx & 7;
      float4 av = *(const float4*)(Ab + (size_t)(mBlock + row) * H_ + k0 + c4 * 4);
      uint2 pv;
      pv.x = cvt_pk_bf16(av.x, av.y);
      pv.y = cvt_pk_bf16(av.z, av.w);
      *(uint2*)&As[row * A_STRIDE + c4 * 4] = pv;
    }
    // ---- stage B: 32x128 f32 -> pair-packed bf16 LDS
    #pragma unroll
    for (int i = 0; i < 2; ++i) {
      int idx = tid + i * 256;           // 512 pair chunks
      int k2 = idx >> 5, c4 = idx & 31;
      const float* s0 = Wb + (size_t)(k0 + 2 * k2) * F_ + nBlock + c4 * 4;
      float4 r0 = *(const float4*)s0;
      float4 r1 = *(const float4*)(s0 + F_);
      uint4 pk;
      pk.x = cvt_pk_bf16(r0.x, r1.x);
      pk.y = cvt_pk_bf16(r0.y, r1.y);
      pk.z = cvt_pk_bf16(r0.z, r1.z);
      pk.w = cvt_pk_bf16(r0.w, r1.w);
      *(uint4*)&Bs[k2 * B_STRIDE + c4 * 4] = pk;
    }
    if (k0 + 32 < H_)  // prefetch next weight K-slice (global_prefetch_b8)
      __builtin_prefetch(Wb + (size_t)(k0 + 32 + (tid >> 3)) * F_ + nBlock + (tid & 7) * 16, 0, 1);
    __syncthreads();

    // ---- compute
    FragCast bfr[2];
    #pragma unroll
    for (int nf = 0; nf < 2; ++nf) {
      int col = waveN + nf * 16 + l15;
      #pragma unroll
      for (int v = 0; v < 8; ++v)
        ((unsigned*)&bfr[nf].u)[v] = Bs[(half16 * 4 + v + (v & 4)) * B_STRIDE + col];
    }
    #pragma unroll
    for (int mf = 0; mf < 4; ++mf) {
      int row = waveM + mf * 16 + l15;
      FragCast af;
      const unsigned short* ap = &As[row * A_STRIDE + half16 * 8];
      ((uint4*)&af.u)[0] = *(const uint4*)ap;
      ((uint4*)&af.u)[1] = *(const uint4*)(ap + 16);
      #pragma unroll
      for (int nf = 0; nf < 2; ++nf)
        acc[mf][nf] = __builtin_amdgcn_wmma_f32_16x16x32_bf16(
            false, af.b, false, bfr[nf].b, (short)0, acc[mf][nf], false, false);
    }
  }

  // ---- epilogue: +b1, gelu, bf16 store into hbuf[b]
  unsigned short* hout = hbuf + (size_t)b * S_ * F_;
  #pragma unroll
  for (int mf = 0; mf < 4; ++mf) {
    #pragma unroll
    for (int nf = 0; nf < 2; ++nf) {
      int gn = nBlock + waveN + nf * 16 + l15;
      float bias = b1e[gn];
      int gm0 = mBlock + waveM + mf * 16 + half16 * 8;
      unsigned short* dst = hout + (size_t)gm0 * F_ + gn;
      #pragma unroll
      for (int r = 0; r < 8; ++r) {
        float v = acc[mf][nf][r] + bias;
        dst[(size_t)r * F_] = f2bf(gelu_tanh(v));
      }
    }
  }
}

// GEMM2: hbuf[b] (bf16 [S_,F_]) x w2[e] (f32 [F_,H_]) + b2 -> f32 out
__global__ void gemm2_kernel(const unsigned short* __restrict__ hbuf,
                             const float* __restrict__ w2,
                             const float* __restrict__ b2,
                             const int* __restrict__ choice,
                             float* __restrict__ out) {
  __shared__ unsigned short As[128 * A_STRIDE];
  __shared__ unsigned       Bs[16 * B_STRIDE];

  const int b = blockIdx.z;
  const int e = choice[b];
  const int mBlock = blockIdx.x * 128;   // over S_
  const int nBlock = blockIdx.y * 128;   // over H_
  const int tid  = threadIdx.x;
  const int lane = tid & 31;
  const int l15  = lane & 15;
  const int half16 = (lane < 16) ? 0 : 1;
  const int wid  = tid >> 5;
  const int waveM = (wid & 1) * 64;
  const int waveN = (wid >> 1) * 32;

  const unsigned short* Ab = hbuf + (size_t)b * S_ * F_;
  const float* Wb  = w2 + (size_t)e * F_ * H_;
  const float* b2e = b2 + (size_t)e * H_;

  v8f acc[4][2];
  #pragma unroll
  for (int mf = 0; mf < 4; ++mf)
    #pragma unroll
    for (int nf = 0; nf < 2; ++nf)
      acc[mf][nf] = (v8f)0.0f;

  for (int k0 = 0; k0 < F_; k0 += 32) {
    __syncthreads();
    // ---- stage A: 128x32 bf16 pure copy via async global->LDS (ASYNCcnt)
    #pragma unroll
    for (int i = 0; i < 2; ++i) {
      int idx = tid + i * 256;           // 512 x 16B chunks (8 bf16 each)
      int row = idx >> 2, c8 = idx & 3;
      const void* g = (const void*)(Ab + (size_t)(mBlock + row) * F_ + k0 + c8 * 8);
      unsigned lds_off = (unsigned)(uintptr_t)&As[row * A_STRIDE + c8 * 8];
      async_copy_b128(lds_off, g);
    }
    // ---- stage B: 32x128 f32 -> pair-packed bf16 LDS (overlaps async copy)
    #pragma unroll
    for (int i = 0; i < 2; ++i) {
      int idx = tid + i * 256;
      int k2 = idx >> 5, c4 = idx & 31;
      const float* s0 = Wb + (size_t)(k0 + 2 * k2) * H_ + nBlock + c4 * 4;
      float4 r0 = *(const float4*)s0;
      float4 r1 = *(const float4*)(s0 + H_);
      uint4 pk;
      pk.x = cvt_pk_bf16(r0.x, r1.x);
      pk.y = cvt_pk_bf16(r0.y, r1.y);
      pk.z = cvt_pk_bf16(r0.z, r1.z);
      pk.w = cvt_pk_bf16(r0.w, r1.w);
      *(uint4*)&Bs[k2 * B_STRIDE + c4 * 4] = pk;
    }
    if (k0 + 32 < F_)
      __builtin_prefetch(Wb + (size_t)(k0 + 32 + (tid >> 3)) * H_ + nBlock + (tid & 7) * 16, 0, 1);
    wait_asynccnt0();   // own-wave async copies complete; barrier orders others
    __syncthreads();

    // ---- compute
    FragCast bfr[2];
    #pragma unroll
    for (int nf = 0; nf < 2; ++nf) {
      int col = waveN + nf * 16 + l15;
      #pragma unroll
      for (int v = 0; v < 8; ++v)
        ((unsigned*)&bfr[nf].u)[v] = Bs[(half16 * 4 + v + (v & 4)) * B_STRIDE + col];
    }
    #pragma unroll
    for (int mf = 0; mf < 4; ++mf) {
      int row = waveM + mf * 16 + l15;
      FragCast af;
      const unsigned short* ap = &As[row * A_STRIDE + half16 * 8];
      ((uint4*)&af.u)[0] = *(const uint4*)ap;
      ((uint4*)&af.u)[1] = *(const uint4*)(ap + 16);
      #pragma unroll
      for (int nf = 0; nf < 2; ++nf)
        acc[mf][nf] = __builtin_amdgcn_wmma_f32_16x16x32_bf16(
            false, af.b, false, bfr[nf].b, (short)0, acc[mf][nf], false, false);
    }
  }

  // ---- epilogue: +b2, f32 store
  float* obase = out + (size_t)b * S_ * H_;
  #pragma unroll
  for (int mf = 0; mf < 4; ++mf) {
    #pragma unroll
    for (int nf = 0; nf < 2; ++nf) {
      int gn = nBlock + waveN + nf * 16 + l15;
      float bias = b2e[gn];
      int gm0 = mBlock + waveM + mf * 16 + half16 * 8;
      float* dst = obase + (size_t)gm0 * H_ + gn;
      #pragma unroll
      for (int r = 0; r < 8; ++r)
        dst[(size_t)r * H_] = acc[mf][nf][r] + bias;
    }
  }
}

// ---------------------------------------------------------------------------
extern "C" void kernel_launch(void* const* d_in, const int* in_sizes, int n_in,
                              void* d_out, int out_size, void* d_ws, size_t ws_size,
                              hipStream_t stream) {
  (void)in_sizes; (void)n_in; (void)out_size; (void)ws_size;
  const float* hidden = (const float*)d_in[0];   // [B,S,H]
  const float* gate_w = (const float*)d_in[1];   // [E,H]
  const float* w1     = (const float*)d_in[2];   // [E,H,F]
  const float* b1     = (const float*)d_in[3];   // [E,F]
  const float* w2     = (const float*)d_in[4];   // [E,F,H]
  const float* b2     = (const float*)d_in[5];   // [E,H]

  float* out = (float*)d_out;                          // [B,S,H] ++ [B,E]
  float* logits_out = out + (size_t)B_ * S_ * H_;

  int* choice = (int*)d_ws;                            // [B_] ints
  unsigned short* hbuf = (unsigned short*)((char*)d_ws + 256);  // bf16 [B,S,F] (96 MB)

  router_kernel<<<B_, 256, 0, stream>>>(hidden, gate_w, logits_out, choice);
  gemm1_kernel<<<dim3(S_ / 128, F_ / 128, B_), 256, 0, stream>>>(hidden, w1, b1, choice, hbuf);
  gemm2_kernel<<<dim3(S_ / 128, H_ / 128, B_), 256, 0, stream>>>(hbuf, w2, b2, choice, out);
}
